// PromptedAttention_4595615007011
// MI455X (gfx1250) — compile-verified, matrix operand order checked by reading, not run on hardware
//
#include <hip/hip_runtime.h>
#include <stdint.h>
#include <math.h>

// ---------------------------------------------------------------------------
// PromptedAttention forward for MI455X (gfx1250, wave32, WMMA bf16).
// B=8, N=1024, C=768, H=12, hd=64, P=50 -> KV length NP=1074.
// Outputs: out [8192,768] f32 followed by attn [96,1024,1074] f32.
//
// Pipeline:
//   1) qkv = x @ Wqkv                          (tiled bf16 WMMA GEMM)
//   2) fused QK^T -> masked softmax -> attn@V  (attn written to HBM ONCE)
//   3) out = ctx @ Wproj + bproj               (tiled bf16 WMMA GEMM)
// ---------------------------------------------------------------------------

typedef __attribute__((ext_vector_type(16))) __bf16       v16bf;
typedef __attribute__((ext_vector_type(2)))  __bf16       v2bf;
typedef __attribute__((ext_vector_type(8)))  float        v8f;
typedef __attribute__((ext_vector_type(4)))  float        f32x4;
typedef __attribute__((ext_vector_type(4)))  unsigned int u32x4;

#define NP    1074   // N + P
#define PHALF 50
#define CDIM  768
#define QKVLD 2304
#define NTOK  1024
#define HEADS 12

union FragBF  { v16bf v; u32x4    q[2]; };
union FragBF2 { v16bf v; uint32_t d[8]; };

// f32 pair -> packed bf16 dword (round-to-nearest-even)
__device__ __forceinline__ uint32_t bf16pack(float a, float b) {
#if __has_builtin(__builtin_amdgcn_cvt_pk_bf16_f32)
  union { v2bf v; uint32_t u; } c;
  c.v = __builtin_amdgcn_cvt_pk_bf16_f32(a, b);
  return c.u;
#else
  union { float f; uint32_t u; } ca, cb;
  ca.f = a; cb.f = b;
  uint32_t ua = (ca.u + 0x7FFFu + ((ca.u >> 16) & 1u)) >> 16;
  uint32_t ub = (cb.u + 0x7FFFu + ((cb.u >> 16) & 1u)) >> 16;
  return (ua & 0xFFFFu) | (ub << 16);
#endif
}
__device__ __forceinline__ uint16_t bf16one(float a) {
  return (uint16_t)(bf16pack(a, 0.f) & 0xFFFFu);
}

// Load one 16x32 bf16 WMMA operand fragment from LDS laid out row-major
// (row = M for A, row = N for transposed-B), row stride `ldw` dwords, k-window
// starting at dword `koff`. Lane layout per ISA: lanes 0-15 take
// k=[0..7]+[16..23], lanes 16-31 take k=[8..15]+[24..31].
__device__ __forceinline__ v16bf load_frag(const uint32_t* lds, int row, int ldw,
                                           int koff, int half) {
  const uint32_t* p = lds + row * ldw + koff + (half << 2);
  FragBF f;
  f.q[0] = *(const u32x4*)(p);
  f.q[1] = *(const u32x4*)(p + 8);
  return f.v;
}

// Assemble a 16x32 bf16 operand fragment for this lane directly from a global
// f32 row (64B-aligned). The runtime `half` goes into the ADDRESS, all packing
// uses compile-time indices (no dynamically-indexed register arrays!).
__device__ __forceinline__ v16bf frag_from_row(const float* row, int half) {
  const float* pk = row + half * 8;
  const f32x4 t0 = *(const f32x4*)(pk);
  const f32x4 t1 = *(const f32x4*)(pk + 4);
  const f32x4 t2 = *(const f32x4*)(pk + 16);
  const f32x4 t3 = *(const f32x4*)(pk + 20);
  FragBF2 f;
  f.d[0] = bf16pack(t0.x, t0.y); f.d[1] = bf16pack(t0.z, t0.w);
  f.d[2] = bf16pack(t1.x, t1.y); f.d[3] = bf16pack(t1.z, t1.w);
  f.d[4] = bf16pack(t2.x, t2.y); f.d[5] = bf16pack(t2.z, t2.w);
  f.d[6] = bf16pack(t3.x, t3.y); f.d[7] = bf16pack(t3.z, t3.w);
  return f.v;
}

__device__ __forceinline__ v8f wmma_bf16(v16bf a, v16bf b, v8f c) {
  return __builtin_amdgcn_wmma_f32_16x16x32_bf16(
      /*neg_a=*/false, a, /*neg_b=*/false, b,
      /*c_mod=*/(short)0, c, /*reuse_a=*/false, /*reuse_b=*/false);
}

// ---------------------------------------------------------------------------
// Generic C = A[MxK] * B[KxN] (+bias) with f32 in/out, bf16 WMMA compute.
// Tile 128x128 per block (256 thr = 8 waves; wave tile 32x64 = 2x4 accums).
// Requires M%128==0, N%128==0, K%32==0 (true for all uses here).
// ---------------------------------------------------------------------------
template <int BIAS>
__global__ __launch_bounds__(256) void gemm_bf16_wmma(
    const float* __restrict__ A, const float* __restrict__ B,
    const float* __restrict__ bias, float* __restrict__ C,
    int M, int N, int K) {
  __shared__ uint32_t sA[128 * 20];  // 128 rows x 32 bf16 (+pad), row = M
  __shared__ uint32_t sB[128 * 20];  // transposed: 128 rows x 32 bf16, row = N
  const int tid  = threadIdx.x;
  const int lane = tid & 31, wid = tid >> 5;
  const int wm = wid & 3, wn = wid >> 2;          // 4 waves in M, 2 in N
  const int mBase = blockIdx.y * 128;
  const int nBase = blockIdx.x * 128;

  v8f acc[2][4] = {};

  for (int k0 = 0; k0 < K; k0 += 32) {
    {  // A tile: coalesced b128 reads
      const int r = tid >> 1, cs = tid & 1;
      const float* src = A + (size_t)(mBase + r) * K + k0 + cs * 16;
      if (k0 + 32 < K) __builtin_prefetch(src + 32, 0, 0);  // next K-tile
      const f32x4* s4 = (const f32x4*)src;
      uint32_t* dst = sA + r * 20 + cs * 8;
#pragma unroll
      for (int j = 0; j < 4; j++) {
        f32x4 t = s4[j];
        dst[2 * j]     = bf16pack(t.x, t.y);
        dst[2 * j + 1] = bf16pack(t.z, t.w);
      }
    }
    {  // B tile, stored transposed [n][k]
      const int n = tid & 127, ks = tid >> 7;
      const float* src = B + (size_t)(k0 + ks * 16) * N + nBase + n;
      float f[16];
#pragma unroll
      for (int j = 0; j < 16; j++) f[j] = src[(size_t)j * N];
      uint32_t* dst = sB + n * 20 + ks * 8;
#pragma unroll
      for (int j = 0; j < 8; j++) dst[j] = bf16pack(f[2 * j], f[2 * j + 1]);
    }
    __syncthreads();

    const int half = lane >> 4, rsel = lane & 15;
    v16bf af[2], bfr[4];
#pragma unroll
    for (int ti = 0; ti < 2; ti++)
      af[ti] = load_frag(sA, wm * 32 + ti * 16 + rsel, 20, 0, half);
#pragma unroll
    for (int tj = 0; tj < 4; tj++)
      bfr[tj] = load_frag(sB, wn * 64 + tj * 16 + rsel, 20, 0, half);
#pragma unroll
    for (int ti = 0; ti < 2; ti++)
#pragma unroll
      for (int tj = 0; tj < 4; tj++)
        acc[ti][tj] = wmma_bf16(af[ti], bfr[tj], acc[ti][tj]);
    __syncthreads();
  }

  const int half = lane >> 4, nlo = lane & 15;
#pragma unroll
  for (int ti = 0; ti < 2; ti++)
#pragma unroll
    for (int tj = 0; tj < 4; tj++) {
      const int col = nBase + wn * 64 + tj * 16 + nlo;
      const float bv = BIAS ? bias[col] : 0.f;
#pragma unroll
      for (int i = 0; i < 8; i++) {
        const int row = mBase + wm * 32 + ti * 16 + half * 8 + i;
        C[(size_t)row * N + col] = acc[ti][tj][i] + bv;
      }
    }
}

// ---------------------------------------------------------------------------
// Fused QK^T -> masked softmax -> attn@V for one (b,h) and 16 query rows.
// attn is written to HBM exactly once; the attn@V pass consumes the bf16
// copy held in LDS. 256 threads = 8 waves:
//   QK/softmax: wave w owns 9 strided 16-col tiles (t = w, w+8, ..., w+64).
//   attn@V:     wave w owns d-tile (w&3) and K-steps s = (w>>2), +2, ...
// ---------------------------------------------------------------------------
__global__ __launch_bounds__(256) void attn_fused_kernel(
    const float* __restrict__ qkv, const float* __restrict__ prompts,
    const unsigned char* __restrict__ mask,
    float* __restrict__ attn, float* __restrict__ ctx) {
  __shared__ uint32_t sQ[16 * 36];      // 16 x 64 bf16 (+pad)
  __shared__ uint32_t sAttn[16 * 584];  // 16 x 1152 bf16 (+pad), LDW=584 dw
  __shared__ float    sRed[8][16];
  __shared__ float    sOut[16 * 64];

  const int tid  = threadIdx.x, lane = tid & 31, w = tid >> 5;
  const int half = lane >> 4, nlo = lane & 15;
  const int qBase = blockIdx.x * 16;
  const int bh = blockIdx.y;
  const int b = bh / HEADS, h = bh % HEADS;

  // ---- stage Q stripe (16 x 64) f32 -> bf16 in LDS ----
  {
    const int r = tid >> 4, seg = tid & 15;  // 4 floats per thread
    const f32x4 t = *(const f32x4*)(
        qkv + (size_t)(b * NTOK + qBase + r) * QKVLD + h * 64 + seg * 4);
    uint32_t* dst = sQ + r * 36 + seg * 2;
    dst[0] = bf16pack(t.x, t.y);
    dst[1] = bf16pack(t.z, t.w);
  }
  __syncthreads();

  v16bf qf[2];
#pragma unroll
  for (int ks = 0; ks < 2; ks++) qf[ks] = load_frag(sQ, nlo, 36, ks * 16, half);

  // ---- QK^T: 9 column tiles per wave, K fragments straight from global ----
  v8f acc[9];
#pragma unroll
  for (int j = 0; j < 9; j++) acc[j] = {};
  bool mok[9];

#pragma unroll
  for (int j = 0; j < 9; j++) {
    const int t  = w + 8 * j;        // n-tile 0..71
    const int kk = 16 * t + nlo;     // this lane's K row / logit column
    const bool kvalid = kk < NP;
    mok[j] = kvalid && (mask[(size_t)b * NP + (kvalid ? kk : 0)] != 0);
    const float* krow = (kk < PHALF)
        ? prompts + (size_t)kk * CDIM + h * 64
        : qkv + (size_t)(b * NTOK + kk - PHALF) * QKVLD + CDIM + h * 64;
#pragma unroll
    for (int ks = 0; ks < 2; ks++) {
      v16bf kf;
      if (kvalid) {
        kf = frag_from_row(krow + 32 * ks, half);
      } else {
        FragBF2 z;
#pragma unroll
        for (int jj = 0; jj < 8; jj++) z.d[jj] = 0u;
        kf = z.v;
      }
      acc[j] = wmma_bf16(qf[ks], kf, acc[j]);
    }
  }

  // ---- softmax over full rows (C-layout: row = i + 8*half, col = nlo) ----
  const float scale = 0.125f;  // hd^-0.5
  v8f vmax;
#pragma unroll
  for (int i = 0; i < 8; i++) vmax[i] = -INFINITY;
#pragma unroll
  for (int j = 0; j < 9; j++)
#pragma unroll
    for (int i = 0; i < 8; i++) {
      const float s = mok[j] ? acc[j][i] * scale : -INFINITY;
      vmax[i] = fmaxf(vmax[i], s);
    }
#pragma unroll
  for (int m = 1; m < 16; m <<= 1)
#pragma unroll
    for (int i = 0; i < 8; i++)
      vmax[i] = fmaxf(vmax[i], __shfl_xor(vmax[i], m, 32));
  if (nlo == 0) {
#pragma unroll
    for (int i = 0; i < 8; i++) sRed[w][i + 8 * half] = vmax[i];
  }
  __syncthreads();
  float gmax[8];
#pragma unroll
  for (int i = 0; i < 8; i++) {
    const int row = i + 8 * half;
    float m = sRed[0][row];
#pragma unroll
    for (int ww = 1; ww < 8; ww++) m = fmaxf(m, sRed[ww][row]);
    gmax[i] = m;
  }
  __syncthreads();

  float p[9][8];
  v8f vsum = {};
#pragma unroll
  for (int j = 0; j < 9; j++)
#pragma unroll
    for (int i = 0; i < 8; i++) {
      const float e = mok[j] ? __expf(acc[j][i] * scale - gmax[i]) : 0.f;
      p[j][i] = e;
      vsum[i] += e;
    }
#pragma unroll
  for (int m = 1; m < 16; m <<= 1)
#pragma unroll
    for (int i = 0; i < 8; i++) vsum[i] += __shfl_xor(vsum[i], m, 32);
  if (nlo == 0) {
#pragma unroll
    for (int i = 0; i < 8; i++) sRed[w][i + 8 * half] = vsum[i];
  }
  __syncthreads();
  float ginv[8];
#pragma unroll
  for (int i = 0; i < 8; i++) {
    const int row = i + 8 * half;
    float s = sRed[0][row];
#pragma unroll
    for (int ww = 1; ww < 8; ww++) s += sRed[ww][row];
    ginv[i] = 1.f / s;
  }

  // ---- write attn (f32, once) + bf16 stripe to LDS for the AV pass ----
  uint16_t* sAttnH = (uint16_t*)sAttn;
#pragma unroll
  for (int j = 0; j < 9; j++) {
    const int c = 16 * (w + 8 * j) + nlo;
#pragma unroll
    for (int i = 0; i < 8; i++) {
      const int row = i + 8 * half;
      const float a = p[j][i] * ginv[i];
      if (c < NP)
        attn[((size_t)bh * NTOK + qBase + row) * NP + c] = a;
      sAttnH[row * 1168 + c] = bf16one(a);  // padded cols get 0
    }
  }
  __syncthreads();

  // ---- attn @ V: wave -> (d-tile, k-parity); 36 K-steps of 32 total ----
  const int dt  = w & 3;           // d-tile (16 cols of hd)
  const int ks0 = w >> 2;          // k-step parity
  const int dcol = dt * 16 + nlo;
  // per-lane column pointers: prompt-V rows and qkv-V rows
  const float* vp = prompts + (size_t)PHALF * CDIM + h * 64 + dcol;  // + (kb-PHALF)*CDIM
  const float* vq = qkv + (size_t)b * NTOK * QKVLD + 2 * CDIM + h * 64 + dcol;  // + (kb-PHALF)*QKVLD
  v8f accv = {};
  for (int s = ks0; s < 36; s += 2) {
    const v16bf afrag = load_frag(sAttn, nlo, 584, s * 16, half);
    FragBF2 vf;
    const int kbase = 32 * s + half * 8;  // lane's first k in this step
    if (kbase >= PHALF && 32 * s + 32 <= NP) {
      // fast path (31 of 36 steps): everything from qkv-v, no compares
      const float* base = vq + (size_t)(kbase - PHALF) * QKVLD;
#pragma unroll
      for (int g = 0; g < 2; g++)
#pragma unroll
        for (int jj = 0; jj < 4; jj++) {
          const float f0 = base[(size_t)(g * 16 + 2 * jj) * QKVLD];
          const float f1 = base[(size_t)(g * 16 + 2 * jj + 1) * QKVLD];
          vf.d[g * 4 + jj] = bf16pack(f0, f1);
        }
    } else {
      // boundary steps: prompt/qkv split and NP tail
#pragma unroll
      for (int g = 0; g < 2; g++)
#pragma unroll
        for (int jj = 0; jj < 4; jj++) {
          const int kb = kbase + g * 16 + 2 * jj;
          float f0 = 0.f, f1 = 0.f;
          if (kb < NP)
            f0 = (kb < PHALF) ? vp[(size_t)(kb - PHALF) * CDIM]
                              : vq[(size_t)(kb - PHALF) * QKVLD];
          if (kb + 1 < NP)
            f1 = (kb + 1 < PHALF) ? vp[(size_t)(kb + 1 - PHALF) * CDIM]
                                  : vq[(size_t)(kb + 1 - PHALF) * QKVLD];
          vf.d[g * 4 + jj] = bf16pack(f0, f1);
        }
    }
    accv = wmma_bf16(afrag, vf.v, accv);
  }

  // combine the two k-parity partials through LDS, then write ctx
  if (ks0 == 0) {
#pragma unroll
    for (int i = 0; i < 8; i++) sOut[(i + 8 * half) * 64 + dcol] = accv[i];
  }
  __syncthreads();
  if (ks0 == 1) {
#pragma unroll
    for (int i = 0; i < 8; i++) sOut[(i + 8 * half) * 64 + dcol] += accv[i];
  }
  __syncthreads();
  {
    const int r = tid >> 4, d = (tid & 15) * 4;
    const f32x4 vv = *(const f32x4*)(sOut + r * 64 + d);
    *(f32x4*)(ctx + (size_t)(b * NTOK + qBase + r) * CDIM + h * 64 + d) = vv;
  }
}

// ---------------------------------------------------------------------------
extern "C" void kernel_launch(void* const* d_in, const int* in_sizes, int n_in,
                              void* d_out, int out_size, void* d_ws, size_t ws_size,
                              hipStream_t stream) {
  const float*         x       = (const float*)d_in[0];
  const unsigned char* mask    = (const unsigned char*)d_in[1];  // jnp bool = 1B
  const float*         prompts = (const float*)d_in[2];
  const float*         Wqkv    = (const float*)d_in[3];
  const float*         Wproj   = (const float*)d_in[4];
  const float*         bproj   = (const float*)d_in[5];

  float* out  = (float*)d_out;                   // [8192, 768]
  float* attn = out + (size_t)8192 * CDIM;       // [96, 1024, 1074]

  float* qkv = (float*)d_ws;                     // [8192, 2304]  (75.5 MB)
  float* ctx = qkv + (size_t)8192 * QKVLD;       // [8192, 768]   (25.2 MB)

  // 1) qkv = x @ Wqkv                 M=8192 N=2304 K=768
  gemm_bf16_wmma<0><<<dim3(2304 / 128, 8192 / 128), 256, 0, stream>>>(
      x, Wqkv, nullptr, qkv, 8192, 2304, CDIM);

  // 2) fused QK^T -> softmax -> attn@V   (64 q-stripes x 96 (b,h))
  attn_fused_kernel<<<dim3(NTOK / 16, 8 * HEADS), 256, 0, stream>>>(
      qkv, prompts, mask, attn, ctx);

  // 3) out = ctx @ Wproj + bproj      M=8192 N=768 K=768
  gemm_bf16_wmma<1><<<dim3(768 / 128, 8192 / 128), 256, 0, stream>>>(
      ctx, Wproj, bproj, out, 8192, CDIM, CDIM);
}
